// KeyPointLearnerGAT_85280870629873
// MI455X (gfx1250) — matrix operand order repaired and checked by reading, not compiled, not run on hardware
//
#include <hip/hip_runtime.h>
#include <hip/hip_bf16.h>
#include <cstddef>

// ---------------------------------------------------------------------------
// Types for CDNA5 WMMA (wave32): D(f32 16x16) = A(f16 16x32) x B(f16 32x16) + C
// ---------------------------------------------------------------------------
typedef __attribute__((ext_vector_type(16))) _Float16 v16h;
typedef __attribute__((ext_vector_type(8)))  float    v8f;
typedef __attribute__((ext_vector_type(4)))  int      v4i;

#define PX 40          // LDS row pitch (halfs) for 32-wide tiles (16B aligned, bank-skewed)
#define KP 704         // MLP K padded: 26*26=676 -> 704 (22 * 32)
#define NGAT 26
#define BPITCH 40      // LDS pitch (halfs) for staged W1 K-slice rows (32 halfs + skew)

// gfx1250 async copy-to-LDS path (ASYNCcnt), guarded so compile never breaks.
#if defined(__HIP_DEVICE_COMPILE__) && __has_builtin(__builtin_amdgcn_global_load_async_to_lds_b128)
#define HAVE_ASYNC_LDS 1
#else
#define HAVE_ASYNC_LDS 0
#endif

// Pointer flavors the async builtin expects: int4* in AS1 (global) / AS3 (LDS).
typedef __attribute__((address_space(1))) v4i g_v4i;
typedef __attribute__((address_space(3))) v4i l_v4i;

__device__ __forceinline__ void wait_async_le4() {
#if HAVE_ASYNC_LDS
#if __has_builtin(__builtin_amdgcn_s_wait_asynccnt)
  __builtin_amdgcn_s_wait_asynccnt(4);
#else
  asm volatile("s_wait_asynccnt 0x4" ::: "memory");
#endif
#endif
}
__device__ __forceinline__ void wait_async_le0() {
#if HAVE_ASYNC_LDS
#if __has_builtin(__builtin_amdgcn_s_wait_asynccnt)
  __builtin_amdgcn_s_wait_asynccnt(0);
#else
  asm volatile("s_wait_asynccnt 0x0" ::: "memory");
#endif
#endif
}

// ---------------------------------------------------------------------------
// Fragment helpers (A/B: contiguous k order used by both; D per CDNA5 ISA 7.12.2)
// A 16x32 f16: lane l holds row (l&15); 16 consecutive k starting at kh = (l>>4)*16.
// -> two contiguous b128 loads (row-major src).
// ---------------------------------------------------------------------------
__device__ __forceinline__ v16h load_afrag(const _Float16* base, int pitch,
                                           int row0, int kbase, int lane) {
  int r  = row0 + (lane & 15);
  int kh = kbase + ((lane >> 4) << 4);
  const _Float16* p = base + r * pitch + kh;
  union { uint4 u[2]; v16h v; } t;
  t.u[0] = *(const uint4*)(p);        // k = kh .. kh+7
  t.u[1] = *(const uint4*)(p + 8);    // k = kh+8 .. kh+15
  return t.v;
}

// B 32x16 f16: lane l holds column (l&15); 16 consecutive k starting at (l>>4)*16.
// Source is stored TRANSPOSED (bufT[col][k]) so this is two contiguous b128 loads.
__device__ __forceinline__ v16h load_bfragT(const _Float16* baseT, int pitch,
                                            int col0, int kbase, int lane) {
  int c  = col0 + (lane & 15);
  int kb = kbase + ((lane >> 4) << 4);
  const _Float16* p = baseT + c * pitch + kb;
  union { uint4 u[2]; v16h v; } t;
  t.u[0] = *(const uint4*)(p);        // k = kb .. kb+7
  t.u[1] = *(const uint4*)(p + 8);    // k = kb+8 .. kb+15
  return t.v;
}

// D/C 16x16 f32: lane l holds column (l&15), rows mb..mb+7 with mb = 8*(l>>4).
// Store transposed (bufT[n][m]) so the 8 row-values are one contiguous write.
__device__ __forceinline__ void store_dT_f16(_Float16* bufT, int pitch,
                                             int it, int jt, v8f d, int lane) {
  int n  = 16 * jt + (lane & 15);
  int mb = 16 * it + ((lane >> 4) << 3);
  union { _Float16 h[8]; uint4 u; } t;
#pragma unroll
  for (int v = 0; v < 8; ++v) t.h[v] = (_Float16)d[v];
  *(uint4*)(bufT + n * pitch + mb) = t.u;   // one ds_store_b128
}

__device__ __forceinline__ void store_dT_f32(float* bufT, int pitch,
                                             int it, int jt, v8f d, int lane) {
  int n  = 16 * jt + (lane & 15);
  int mb = 16 * it + ((lane >> 4) << 3);
  float* p = bufT + n * pitch + mb;
  float4 lo = {d[0], d[1], d[2], d[3]};
  float4 hi = {d[4], d[5], d[6], d[7]};
  *(float4*)(p)     = lo;
  *(float4*)(p + 4) = hi;
}

__device__ __forceinline__ v8f wmma_f16(v16h a, v16h b, v8f c) {
  return __builtin_amdgcn_wmma_f32_16x16x32_f16(false, a, false, b,
                                                (short)0, c, false, false);
}

// ---------------------------------------------------------------------------
// Prep 1: augmented transposed GAT weights.
// WaugT[l][h][c][k] (f16, 32x32):  c<26 -> W[l][h][k][c]
//                                  c==26 -> (W @ a_src)[k],  c==27 -> (W @ a_dst)[k]
// so one GEMM x@Waug yields h plus both attention score vectors in cols 26/27.
// ---------------------------------------------------------------------------
__global__ void prep_waug(const float* __restrict__ W,
                          const float* __restrict__ a_src,
                          const float* __restrict__ a_dst,
                          _Float16* __restrict__ WaugT) {
  int lh = blockIdx.x;             // 0..11  (l*4 + h)
  int k  = threadIdx.x;            // 0..31  (row of Waug = input-feature index)
  const float* Wm = W + (size_t)lh * NGAT * NGAT;   // W[l][h][k][c]
  const float* as = a_src + lh * NGAT;
  const float* ad = a_dst + lh * NGAT;
  _Float16* outT = WaugT + lh * 32 * 32;
  float vsrc = 0.f, vdst = 0.f;
  if (k < NGAT) {
    for (int f = 0; f < NGAT; ++f) {
      float w = Wm[k * NGAT + f];
      vsrc += w * as[f];
      vdst += w * ad[f];
    }
  }
  for (int c = 0; c < 32; ++c) {
    float v = 0.f;
    if (k < NGAT) {
      if (c < NGAT)       v = Wm[k * NGAT + c];
      else if (c == NGAT) v = vsrc;
      else if (c == 27)   v = vdst;
    }
    outT[c * 32 + k] = (_Float16)v;
  }
}

// Prep 2: W1 transposed + K-padded to f16:  W1T[n][k] = W1[k][n], k<676, else 0.
__global__ void prep_w1t(const float* __restrict__ W1, _Float16* __restrict__ W1T) {
  int idx = blockIdx.x * blockDim.x + threadIdx.x;
  if (idx >= 256 * KP) return;
  int n = idx / KP, k = idx % KP;
  W1T[idx] = (_Float16)((k < 676) ? W1[(size_t)k * 256 + n] : 0.f);
}

// ---------------------------------------------------------------------------
// Fused 3-layer GAT. 1 block = 1 batch element, 4 waves = 4 heads, all in LDS.
// Writes final x as f16, row-padded to KP for the MLP GEMM.
// ---------------------------------------------------------------------------
__global__ __launch_bounds__(128) void gat3(const float* __restrict__ x,
                                            const _Float16* __restrict__ WaugT,
                                            _Float16* __restrict__ xflat) {
  __shared__ __align__(16) _Float16 sX[32 * PX];        // current x tile (f16, padded)
  __shared__ __align__(16) _Float16 sHT[4][32 * PX];    // per-head h_aug, transposed
  __shared__ __align__(16) _Float16 sAl[4][32 * PX];    // per-head alpha, row-major
  __shared__ __align__(16) float    sOT[4][32 * PX];    // per-head out, transposed f32

  const int b    = blockIdx.x;
  const int tid  = threadIdx.x;
  const int wave = tid >> 5;      // head
  const int lane = tid & 31;

  // zero padded regions once (rows/cols >=26 stay zero for all layers)
  for (int i = tid; i < 32 * PX; i += 128) sX[i] = (_Float16)0.f;
  for (int i = tid; i < 4 * 32 * PX; i += 128) (&sAl[0][0])[i] = (_Float16)0.f;
  __syncthreads();

  const float* xb = x + (size_t)b * (NGAT * NGAT);
  for (int i = tid; i < NGAT * NGAT; i += 128)
    sX[(i / NGAT) * PX + (i % NGAT)] = (_Float16)xb[i];
  __syncthreads();

  for (int l = 0; l < 3; ++l) {
    const _Float16* wT = WaugT + ((l * 4 + wave) * 32 * 32);
    _Float16* hT = sHT[wave];

    // ---- GEMM 1: h_aug = X(32x32) @ Waug(32x32), 4 WMMA, K=32 in one step ----
    {
      v16h a0 = load_afrag(sX, PX, 0, 0, lane);
      v16h a1 = load_afrag(sX, PX, 16, 0, lane);
      v16h b0 = load_bfragT(wT, 32, 0, 0, lane);
      v16h b1 = load_bfragT(wT, 32, 16, 0, lane);
      v8f z = {};
      v8f d00 = wmma_f16(a0, b0, z);
      v8f d01 = wmma_f16(a0, b1, z);
      v8f d10 = wmma_f16(a1, b0, z);
      v8f d11 = wmma_f16(a1, b1, z);
      store_dT_f16(hT, PX, 0, 0, d00, lane);
      store_dT_f16(hT, PX, 0, 1, d01, lane);
      store_dT_f16(hT, PX, 1, 0, d10, lane);
      store_dT_f16(hT, PX, 1, 1, d11, lane);
    }
    __syncthreads();

    // ---- softmax over neighbors: lane i owns row i ----
    // s_src[i] = hT[26][i], s_dst[j] = hT[27][j] (free columns of augmented GEMM)
    if (lane < NGAT) {
      float ssrc = (float)hT[26 * PX + lane];
      float e[NGAT];
      float mx = -1e30f;
#pragma unroll
      for (int j = 0; j < NGAT; ++j) {
        float v = ssrc + (float)hT[27 * PX + j];
        v = v > 0.f ? v : 0.2f * v;            // leaky_relu 0.2
        e[j] = v;
        mx = fmaxf(mx, v);
      }
      float s = 0.f;
#pragma unroll
      for (int j = 0; j < NGAT; ++j) { e[j] = __expf(e[j] - mx); s += e[j]; }
      float inv = 1.f / s;
#pragma unroll
      for (int j = 0; j < NGAT; ++j)
        sAl[wave][lane * PX + j] = (_Float16)(e[j] * inv);
    }
    __syncthreads();

    // ---- GEMM 2: out = alpha(32x32) @ h(32x32), 4 WMMA ----
    {
      v16h p0 = load_afrag(sAl[wave], PX, 0, 0, lane);
      v16h p1 = load_afrag(sAl[wave], PX, 16, 0, lane);
      v16h q0 = load_bfragT(hT, PX, 0, 0, lane);   // B[k][f] = h[k][f] = hT[f][k]
      v16h q1 = load_bfragT(hT, PX, 16, 0, lane);
      v8f z = {};
      v8f o00 = wmma_f16(p0, q0, z);
      v8f o01 = wmma_f16(p0, q1, z);
      v8f o10 = wmma_f16(p1, q0, z);
      v8f o11 = wmma_f16(p1, q1, z);
      store_dT_f32(sOT[wave], PX, 0, 0, o00, lane);
      store_dT_f32(sOT[wave], PX, 0, 1, o01, lane);
      store_dT_f32(sOT[wave], PX, 1, 0, o10, lane);
      store_dT_f32(sOT[wave], PX, 1, 1, o11, lane);
    }
    __syncthreads();

    // ---- head mean + ELU -> next x ----
    for (int i = tid; i < NGAT * NGAT; i += 128) {
      int n = i / NGAT, f = i % NGAT;             // node n, feature f
      float v = 0.25f * (sOT[0][f * PX + n] + sOT[1][f * PX + n] +
                         sOT[2][f * PX + n] + sOT[3][f * PX + n]);
      v = v > 0.f ? v : (__expf(v) - 1.f);        // ELU
      sX[n * PX + f] = (_Float16)v;
    }
    __syncthreads();
  }

  // ---- emit flattened, K-padded f16 row for the MLP ----
  _Float16* dst = xflat + (size_t)b * KP;
  for (int i = tid; i < KP; i += 128)
    dst[i] = (i < NGAT * NGAT) ? sX[(i / NGAT) * PX + (i % NGAT)] : (_Float16)0.f;
}

// ---------------------------------------------------------------------------
// MLP head: out = leaky_relu(xflat @ W1 + b1) @ W2 + b2.
// 8 waves/block; each wave owns 16 rows x all 256 cols (128 f32 accum VGPRs),
// 22 K-steps x 16 WMMA. The 32x256 W1 K-slice is staged once per block into a
// double-buffered LDS tile via gfx1250 async copy-to-LDS (ASYNCcnt) when
// available (falls back to synchronous staging), removing the 8x per-wave
// redundant global B traffic. Tiny 256->3 GEMV done on VALU from LDS.
// ---------------------------------------------------------------------------

// Stage one 256x32 slice of W1T (transposed layout) into LDS (pitch BPITCH).
// 1024 16-byte chunks over 256 threads -> 4 async b128 copies per thread.
__device__ __forceinline__ void stage_w1(const _Float16* __restrict__ W1T,
                                         _Float16* dstbuf, int kb, int tid) {
#pragma unroll
  for (int q = 0; q < 4; ++q) {
    int chunk = tid + 256 * q;             // 0..1023
    int c  = chunk >> 2;                   // row (output-channel) 0..255
    int kq = (chunk & 3) << 3;             // 0,8,16,24 halfs within the slice
    const _Float16* src = W1T + (size_t)c * KP + kb + kq;
    _Float16* dst = dstbuf + c * BPITCH + kq;
#if HAVE_ASYNC_LDS
    __builtin_amdgcn_global_load_async_to_lds_b128((g_v4i*)src, (l_v4i*)dst, 0, 0);
#else
    *(uint4*)dst = *(const uint4*)src;
#endif
  }
}

__global__ __launch_bounds__(256) void mlp(const _Float16* __restrict__ A,
                                           const _Float16* __restrict__ W1T,
                                           const float* __restrict__ b1,
                                           const float* __restrict__ W2,
                                           const float* __restrict__ b2,
                                           float* __restrict__ out) {
  __shared__ __align__(16) _Float16 sBT[2][256 * BPITCH];  // double-buffered W1 slice
  __shared__ __align__(16) _Float16 sHT[8][256 * 16];      // per-wave hdn, transposed
  __shared__ float sW2[256 * 3];

  const int tid  = threadIdx.x;
  const int wave = tid >> 5;
  const int lane = tid & 31;

  for (int i = tid; i < 256 * 3; i += 256) sW2[i] = W2[i];

  const int M0 = (blockIdx.x * 8 + wave) * 16;
  const _Float16* arow = A + (size_t)M0 * KP;

  v8f acc[16];
#pragma unroll
  for (int t = 0; t < 16; ++t) acc[t] = (v8f){};

  stage_w1(W1T, sBT[0], 0, tid);           // prologue: stage K-step 0

  for (int ks = 0; ks < 22; ++ks) {
    const int kb  = ks * 32;
    const int cur = ks & 1;
    if (ks + 1 < 22) {                     // overlap: stage next slice
      stage_w1(W1T, sBT[cur ^ 1], kb + 32, tid);
      wait_async_le4();                    // step-ks slice complete (in-order)
    } else {
      wait_async_le0();
    }
    __syncthreads();                       // slice visible to all waves

    if (ks + 1 < 22)                       // pull next A slice (global_prefetch_b8)
      __builtin_prefetch(arow + (lane & 15) * KP + kb + 32, 0, 0);
    v16h af = load_afrag(arow, KP, 0, kb, lane);
#pragma unroll
    for (int t = 0; t < 16; ++t) {
      v16h bf = load_bfragT(sBT[cur], BPITCH, 16 * t, 0, lane);
      acc[t] = wmma_f16(af, bf, acc[t]);
    }
    __syncthreads();                       // done reading before it is re-staged
  }

  // bias + leaky_relu, stash transposed f16 in LDS (one b128 per tile)
#pragma unroll
  for (int t = 0; t < 16; ++t) {
    int n  = 16 * t + (lane & 15);
    int mb = (lane >> 4) << 3;
    float bn = b1[n];
    union { _Float16 h[8]; uint4 u; } pk;
#pragma unroll
    for (int v = 0; v < 8; ++v) {
      float hv = acc[t][v] + bn;
      hv = hv > 0.f ? hv : 0.2f * hv;
      pk.h[v] = (_Float16)hv;
    }
    *(uint4*)(&sHT[wave][n * 16 + mb]) = pk.u;
  }
  __syncthreads();

  // 256 -> 3 GEMV: 48 (row, out-col) tasks per wave
  for (int task = lane; task < 48; task += 32) {
    int m = task & 15;
    int c = task >> 4;
    float s = b2[c];
    for (int n = 0; n < 256; ++n)
      s += (float)sHT[wave][n * 16 + m] * sW2[n * 3 + c];
    out[(size_t)(M0 + m) * 3 + c] = s;
  }
}

// ---------------------------------------------------------------------------
// Launch: prep -> fused GAT -> MLP, all on `stream` (graph-capture safe).
// Workspace: [WaugT 24KB][W1T 352KB][xflat 22MB]
// ---------------------------------------------------------------------------
extern "C" void kernel_launch(void* const* d_in, const int* in_sizes, int n_in,
                              void* d_out, int out_size, void* d_ws, size_t ws_size,
                              hipStream_t stream) {
  const float* x     = (const float*)d_in[1];
  const float* W     = (const float*)d_in[2];
  const float* a_src = (const float*)d_in[3];
  const float* a_dst = (const float*)d_in[4];
  const float* W1    = (const float*)d_in[5];
  const float* b1    = (const float*)d_in[6];
  const float* W2    = (const float*)d_in[7];
  const float* b2    = (const float*)d_in[8];
  float* out = (float*)d_out;

  char* ws = (char*)d_ws;
  _Float16* WaugT = (_Float16*)ws;                               // 12*32*32*2   = 24576 B
  _Float16* W1T   = (_Float16*)(ws + 24576);                     // 256*704*2    = 360448 B
  _Float16* xflat = (_Float16*)(ws + 24576 + 360448);            // 16384*704*2  = ~22 MB

  prep_waug<<<12, 32, 0, stream>>>(W, a_src, a_dst, WaugT);
  prep_w1t<<<(256 * KP + 255) / 256, 256, 0, stream>>>(W1, W1T);
  gat3<<<16384, 128, 0, stream>>>(x, WaugT, xflat);
  mlp<<<16384 / 128, 256, 0, stream>>>(xflat, W1T, b1, W2, b2, out);
}
